// RoPE_MHA_34076270527162
// MI455X (gfx1250) — matrix operand order repaired and checked by reading, not compiled, not run on hardware
//
#include <hip/hip_runtime.h>
#include <math.h>

#define B_     2
#define N_     2048
#define D_     64
#define H_     8
#define INNER_ 512
#define QKV_   1536

typedef __attribute__((ext_vector_type(16))) _Float16 v16h;
typedef __attribute__((ext_vector_type(8)))  _Float16 v8h;
typedef __attribute__((ext_vector_type(8)))  float    v8f;

union AFrag { v16h v; v8h h[2]; };

__device__ __forceinline__ v8f wmma16x16x32(v16h a, v16h b, v8f c) {
  // D = A(16x32 f16) * B(32x16 f16) + C(16x16 f32)
  return __builtin_amdgcn_wmma_f32_16x16x32_f16(false, a, false, b, (short)0, c,
                                                false, false);
}

// CDNA5 async copy: 16 bytes per lane, global -> LDS, tracked by ASYNCcnt.
// VDST = LDS byte address VGPR, VADDR = 64-bit global address VGPR pair.
__device__ __forceinline__ void async_ld_b128(unsigned lds_addr,
                                              const _Float16* gptr) {
  asm volatile("global_load_async_to_lds_b128 %0, %1, off"
               :: "v"(lds_addr), "v"(gptr) : "memory");
}

__device__ __forceinline__ void wait_asynccnt0() {
  asm volatile("s_wait_asynccnt 0x0" ::: "memory");
}

// ---------------------------------------------------------------------------
// Kernel 1: qkv = x @ Wqkv^T + bqkv, fused interleaved RoPE on Q/K sections.
// One wave per 16-row x-tile; 4 column tiles (64 cols) per wave.
// Outputs: Qh,Kh as [b,h,n,64] f16 (rope'd); V transposed as Vt [b,h,64,n] f16.
// ---------------------------------------------------------------------------
__global__ __launch_bounds__(32) void qkv_rope_kernel(
    const float* __restrict__ x, const float* __restrict__ Wqkv,
    const float* __restrict__ bqkv, _Float16* __restrict__ Qh,
    _Float16* __restrict__ Kh, _Float16* __restrict__ Vt)
{
  const int lane   = threadIdx.x;
  const int l15    = lane & 15;
  const int half16 = (lane >> 4) & 1;
  const int base0  = half16 * 8;   // C-fragment row offset for this lane half
  const int koff   = half16 * 16;  // B-fragment K offset for this lane half
  const int mt     = blockIdx.x;   // 16-row tile index over B_*N_ rows

  // A fragments (x rows, f32 -> f16), K = 0..31 and 32..63
  const float* xr = x + (size_t)(mt * 16 + l15) * D_;
  v16h a0, a1;
#pragma unroll
  for (int hh = 0; hh < 8; ++hh) {
    a0[hh]     = (_Float16)xr[koff + hh];
    a0[hh + 8] = (_Float16)xr[koff + 8 + hh];
    a1[hh]     = (_Float16)xr[32 + koff + hh];
    a1[hh + 8] = (_Float16)xr[32 + koff + 8 + hh];
  }

  const int bq    = (mt * 16) / N_;        // batch (tile never straddles batches)
  const int nbase = (mt * 16) & (N_ - 1);  // sequence position of tile row 0

  for (int t = 0; t < 4; ++t) {
    const int col = (blockIdx.y * 4 + t) * 16 + l15;  // qkv output column
    // B[k][col] = Wqkv[col][k]  (row-major [1536,64]) -> contiguous per lane
    const float* wr = Wqkv + (size_t)col * D_ + koff;
    v16h b0, b1;
#pragma unroll
    for (int hh = 0; hh < 16; ++hh) {
      b0[hh] = (_Float16)wr[hh];
      b1[hh] = (_Float16)wr[32 + hh];
    }
    v8f c = {};
    c = wmma16x16x32(a0, b0, c);
    c = wmma16x16x32(a1, b1, c);

    const float bias = bqkv[col];
    const int sec  = col / INNER_;        // 0=Q 1=K 2=V (uniform per wave)
    const int cc   = col - sec * INNER_;
    const int head = cc >> 6;
    const int dim  = cc & 63;

    float vals[8];
#pragma unroll
    for (int r = 0; r < 8; ++r) vals[r] = c[r] + bias;

    if (sec < 2) {
      // interleaved rotary: pair partner sits in the adjacent lane (col^1)
      const float e    = (float)(dim & ~1) * (1.0f / 64.0f);
      const float invf = __expf(-e * 9.210340371976184f);  // 10000^-e
      const float sgn  = (dim & 1) ? 1.0f : -1.0f;
#pragma unroll
      for (int r = 0; r < 8; ++r) {
        const float partner = __shfl_xor(vals[r], 1, 32);
        const int   n = nbase + r + base0;  // C row -> sequence position
        float s, cs;
        __sincosf((float)n * invf, &s, &cs);
        vals[r] = vals[r] * cs + sgn * partner * s;
      }
      _Float16* dst = (sec == 0) ? Qh : Kh;
#pragma unroll
      for (int r = 0; r < 8; ++r) {
        const int n = nbase + r + base0;
        dst[(((size_t)(bq * H_ + head)) * N_ + n) * D_ + dim] = (_Float16)vals[r];
      }
    } else {
      // V: store transposed [b,h,d,n]; this lane's 8 rows are contiguous n
      _Float16* vp = Vt + (((size_t)(bq * H_ + head)) * D_ + dim) * N_ + nbase + base0;
#pragma unroll
      for (int r = 0; r < 8; ++r) vp[r] = (_Float16)vals[r];
    }
  }
}

// ---------------------------------------------------------------------------
// Kernel 2: flash attention. Block = 4 waves sharing one (b,h); each wave owns
// a 16-query strip. K/V tiles staged in LDS via GLOBAL_LOAD_ASYNC_TO_LDS_B128,
// double-buffered (copy of tile i+1 overlaps compute of tile i), synchronized
// with s_wait_asynccnt + one workgroup barrier per step. Computes S^T = K*Q^T
// so softmax stats are lane-local and exp(S^T) lands directly in the P
// A-operand layout for P*V.
// ---------------------------------------------------------------------------
#define KSTR 80   // LDS K-tile row stride in halves (64 + pad, 160B: 16B-aligned)
#define VSTR 48   // LDS V-tile row stride in halves (32 + pad,  96B: 16B-aligned)
#define VOFS (32 * KSTR)             // V tile offset inside one buffer (halves)
#define SBUF (32 * KSTR + 64 * VSTR) // halves per buffer

__global__ __launch_bounds__(128) void flash_attn_kernel(
    const _Float16* __restrict__ Qh, const _Float16* __restrict__ Kh,
    const _Float16* __restrict__ Vt, _Float16* __restrict__ attnH)
{
  __shared__ _Float16 smem[2][SBUF];

  const int tid    = threadIdx.x;
  const int lane   = tid & 31;
  const int wave   = tid >> 5;
  const int l15    = lane & 15;
  const int half16 = (lane >> 4) & 1;
  const int base0  = half16 * 8;
  const int koff   = half16 * 16;

  const int bh = blockIdx.x;                 // b*8 + h
  const int b  = bh >> 3;
  const int h  = bh & 7;
  const int n0 = blockIdx.y * 64 + wave * 16;  // query-row base for this wave

  const _Float16* Qp = Qh + (size_t)bh * N_ * D_;
  const _Float16* Kp = Kh + (size_t)bh * N_ * D_;
  const _Float16* Vp = Vt + (size_t)bh * D_ * N_;

  // Loop-invariant B operand for S^T: Bq[d][m] = Q[n0+m][d]
  const v16h bq0 = __builtin_shufflevector(*(const v8h*)(Qp + (size_t)(n0 + l15) * D_ + base0), *(const v8h*)(Qp + (size_t)(n0 + l15) * D_ + base0 + 16), 0, 1, 2, 3, 4, 5, 6, 7, 8, 9, 10, 11, 12, 13, 14, 15);
  const v16h bq1 = __builtin_shufflevector(*(const v8h*)(Qp + (size_t)(n0 + l15) * D_ + 32 + base0), *(const v8h*)(Qp + (size_t)(n0 + l15) * D_ + 32 + base0 + 16), 0, 1, 2, 3, 4, 5, 6, 7, 8, 9, 10, 11, 12, 13, 14, 15);

  v8f o0 = {}, o1 = {}, o2 = {}, o3 = {};   // O tiles: 16 q-rows x 64 dims
  float mrun = -__builtin_inff(), lrun = 0.0f;

  // Cooperative async copy of one 32-key tile: K 32x64 (8 chunks/row) and
  // V^T 64x32 (4 chunks/row); 512 x 16B chunks, 4 per thread.
#define ISSUE_TILE(J0, BUF)                                                    \
  {                                                                            \
    const unsigned lb = (unsigned)(size_t)(&smem[(BUF)][0]);                   \
    _Pragma("unroll")                                                          \
    for (int i = 0; i < 2; ++i) {                                              \
      const int q = tid + i * 128;                                             \
      const int kr = q >> 3, kc = q & 7;                                       \
      async_ld_b128(lb + (unsigned)(kr * KSTR + kc * 8) * 2,                   \
                    Kp + (size_t)((J0) + kr) * D_ + kc * 8);                   \
      const int vr = q >> 2, vc = q & 3;                                       \
      async_ld_b128(lb + (unsigned)(VOFS + vr * VSTR + vc * 8) * 2,            \
                    Vp + (size_t)vr * N_ + (J0) + vc * 8);                     \
    }                                                                          \
  }

  ISSUE_TILE(0, 0);
  int cur = 0;

  for (int j0 = 0; j0 < N_; j0 += 32) {
    wait_asynccnt0();   // this wave's copies into buffer `cur` have landed
    __syncthreads();    // all waves' copies done; prior reads of cur^1 done
    if (j0 + 32 < N_) ISSUE_TILE(j0 + 32, cur ^ 1);

    const _Float16* bufK = &smem[cur][0];
    const _Float16* bufV = &smem[cur][VOFS];

    // A operands from LDS: K rows j0..j0+31 (two j-tiles x two d-chunks)
    const _Float16* kr0 = bufK + (size_t)l15 * KSTR;
    const _Float16* kr1 = kr0 + 16 * KSTR;
    AFrag ak00, ak01, ak10, ak11;
    ak00.h[0] = *(const v8h*)(kr0 + base0);
    ak00.h[1] = *(const v8h*)(kr0 + base0 + 16);
    ak01.h[0] = *(const v8h*)(kr0 + 32 + base0);
    ak01.h[1] = *(const v8h*)(kr0 + 32 + base0 + 16);
    ak10.h[0] = *(const v8h*)(kr1 + base0);
    ak10.h[1] = *(const v8h*)(kr1 + base0 + 16);
    ak11.h[0] = *(const v8h*)(kr1 + 32 + base0);
    ak11.h[1] = *(const v8h*)(kr1 + 32 + base0 + 16);

    // S^T tiles: rows = key index j (vgpr), cols = query m (lane)
    v8f s0 = {}, s1 = {};
    s0 = wmma16x16x32(ak00.v, bq0, s0);
    s0 = wmma16x16x32(ak01.v, bq1, s0);
    s1 = wmma16x16x32(ak10.v, bq0, s1);
    s1 = wmma16x16x32(ak11.v, bq1, s1);

    // Online softmax over j (lane-local + one xor16 merge of row halves)
    float tmax = -__builtin_inff();
#pragma unroll
    for (int r = 0; r < 8; ++r) {
      s0[r] *= 0.125f;  // 1/sqrt(64)
      s1[r] *= 0.125f;
      tmax = fmaxf(tmax, fmaxf(s0[r], s1[r]));
    }
    tmax = fmaxf(tmax, __shfl_xor(tmax, 16, 32));
    const float mnew   = fmaxf(mrun, tmax);
    const float fscale = __expf(mrun - mnew);
    mrun = mnew;

    // exp(S^T) in C layout == P in A layout (halves 0-7 <- St0, 8-15 <- St1)
    AFrag ap;
    float psum = 0.0f;
#pragma unroll
    for (int r = 0; r < 8; ++r) {
      const float p0 = __expf(s0[r] - mnew);
      const float p1 = __expf(s1[r] - mnew);
      psum += p0 + p1;
      ap.v[r]     = (_Float16)p0;
      ap.v[r + 8] = (_Float16)p1;
    }
    psum += __shfl_xor(psum, 16, 32);
    lrun = lrun * fscale + psum;

    // Rescale O: factor for row m fetched from lane m (bpermute)
#pragma unroll
    for (int r = 0; r < 8; ++r) {
      const float fr = __shfl(fscale, r + base0, 32);
      o0[r] *= fr; o1[r] *= fr; o2[r] *= fr; o3[r] *= fr;
    }

    // O += P*V ; Bv[j][d'] = V^T[d'][j] from LDS -> two v8h per lane (j = base0.., base0+16..)
    const _Float16* vb = bufV + (size_t)l15 * VSTR + base0;
    o0 = wmma16x16x32(ap.v, __builtin_shufflevector(*(const v8h*)(vb + 0 * 16 * VSTR), *(const v8h*)(vb + 0 * 16 * VSTR + 16), 0, 1, 2, 3, 4, 5, 6, 7, 8, 9, 10, 11, 12, 13, 14, 15), o0);
    o1 = wmma16x16x32(ap.v, __builtin_shufflevector(*(const v8h*)(vb + 1 * 16 * VSTR), *(const v8h*)(vb + 1 * 16 * VSTR + 16), 0, 1, 2, 3, 4, 5, 6, 7, 8, 9, 10, 11, 12, 13, 14, 15), o1);
    o2 = wmma16x16x32(ap.v, __builtin_shufflevector(*(const v8h*)(vb + 2 * 16 * VSTR), *(const v8h*)(vb + 2 * 16 * VSTR + 16), 0, 1, 2, 3, 4, 5, 6, 7, 8, 9, 10, 11, 12, 13, 14, 15), o2);
    o3 = wmma16x16x32(ap.v, __builtin_shufflevector(*(const v8h*)(vb + 3 * 16 * VSTR), *(const v8h*)(vb + 3 * 16 * VSTR + 16), 0, 1, 2, 3, 4, 5, 6, 7, 8, 9, 10, 11, 12, 13, 14, 15), o3);

    cur ^= 1;
  }

  // Normalize by row sum and store heads-merged [b,n,512] f16
  const float linv = 1.0f / lrun;
#pragma unroll
  for (int r = 0; r < 8; ++r) {
    const float fr   = __shfl(linv, r + base0, 32);
    const int   nrow = n0 + r + base0;
    _Float16* op = attnH + ((size_t)(b * N_ + nrow)) * INNER_ + h * D_ + l15;
    op[0]  = (_Float16)(o0[r] * fr);
    op[16] = (_Float16)(o1[r] * fr);
    op[32] = (_Float16)(o2[r] * fr);
    op[48] = (_Float16)(o3[r] * fr);
  }
}

// ---------------------------------------------------------------------------
// Kernel 3: out = attn @ Wout^T + bout.  One wave per 16x16 output tile,
// K-loop of 16 WMMAs over INNER_=512.
// ---------------------------------------------------------------------------
__global__ __launch_bounds__(32) void out_proj_kernel(
    const _Float16* __restrict__ attnH, const float* __restrict__ Wout,
    const float* __restrict__ bout, float* __restrict__ out)
{
  const int lane   = threadIdx.x;
  const int l15    = lane & 15;
  const int half16 = (lane >> 4) & 1;
  const int base0  = half16 * 8;
  const int koff   = half16 * 16;
  const int row = blockIdx.x * 16 + l15;  // A row (b*n flattened)
  const int col = blockIdx.y * 16 + l15;  // output feature

  const _Float16* ar = attnH + (size_t)row * INNER_;
  const float*    wr = Wout + (size_t)col * INNER_;

  v8f c = {};
  for (int k0 = 0; k0 < INNER_; k0 += 32) {
    AFrag a;
    a.h[0] = *(const v8h*)(ar + k0 + koff);
    a.h[1] = *(const v8h*)(ar + k0 + koff + 8);
    v16h bf;
#pragma unroll
    for (int hh = 0; hh < 16; ++hh) bf[hh] = (_Float16)wr[k0 + koff + hh];
    c = wmma16x16x32(a.v, bf, c);
  }

  const float bias = bout[col];
#pragma unroll
  for (int r = 0; r < 8; ++r) {
    const int m = blockIdx.x * 16 + r + base0;
    out[(size_t)m * D_ + col] = c[r] + bias;
  }
}

// ---------------------------------------------------------------------------
extern "C" void kernel_launch(void* const* d_in, const int* in_sizes, int n_in,
                              void* d_out, int out_size, void* d_ws, size_t ws_size,
                              hipStream_t stream) {
  const float* x    = (const float*)d_in[0];  // q == k == v
  const float* Wqkv = (const float*)d_in[3];
  const float* bqkv = (const float*)d_in[4];
  const float* Wout = (const float*)d_in[5];
  const float* bout = (const float*)d_in[6];

  // Workspace carve-up (f16): Qh 4MB | Kh 4MB | Vt 4MB | attnH 4MB = 16MB
  char* ws = (char*)d_ws;
  _Float16* Qh    = (_Float16*)(ws);
  _Float16* Kh    = (_Float16*)(ws + ((size_t)4 << 20));
  _Float16* Vt    = (_Float16*)(ws + ((size_t)8 << 20));
  _Float16* attnH = (_Float16*)(ws + ((size_t)12 << 20));

  qkv_rope_kernel<<<dim3(B_ * N_ / 16, QKV_ / 64), 32, 0, stream>>>(
      x, Wqkv, bqkv, Qh, Kh, Vt);
  flash_attn_kernel<<<dim3(B_ * H_, N_ / 64), 128, 0, stream>>>(
      Qh, Kh, Vt, attnH);
  out_proj_kernel<<<dim3(B_ * N_ / 16, D_ / 16), 32, 0, stream>>>(
      attnH, Wout, bout, (float*)d_out);
}
